// LearnableLinear_68436008894608
// MI455X (gfx1250) — matrix operand order repaired and checked by reading, not compiled, not run on hardware
//
#include <hip/hip_runtime.h>
#include <stdint.h>

#define IN_DIM   512
#define OUT_DIM  512
#define RANK     4
#define HDIM     4608           // (512+512)*4 + 512
#define G4       (4*HDIM)       // 18432
#define BATCH    256
#define KTOT     (IN_DIM+HDIM)  // 5120
#define KB_TOT   (KTOT/32)      // 160
#define APAD     40             // padded LDS row stride (halves): bank-conflict-free

#define GAS __attribute__((address_space(1)))
#define LAS __attribute__((address_space(3)))

typedef __attribute__((ext_vector_type(16))) __bf16 v16bf;
typedef __attribute__((ext_vector_type(4)))  __bf16 v4bf;
typedef __attribute__((ext_vector_type(2)))  __bf16 v2bf;
typedef __attribute__((ext_vector_type(8)))  float  v8f;
typedef __attribute__((ext_vector_type(4)))  int    v4i;

// packed f32x2 -> bf16x2 (RNE), kept in vector form
static __device__ __forceinline__ v2bf cvt2(float lo, float hi) {
#if __has_builtin(__builtin_amdgcn_cvt_pk_bf16_f32)
  return __builtin_amdgcn_cvt_pk_bf16_f32(lo, hi);
#else
  v2bf r; r.x = (__bf16)lo; r.y = (__bf16)hi; return r;
#endif
}

static __device__ __forceinline__ unsigned short f2bf(float f) {
  __bf16 b = (__bf16)f;
  unsigned short u; __builtin_memcpy(&u, &b, 2);
  return u;
}

// async global(16B) -> LDS copy, tracked by ASYNCcnt
static __device__ __forceinline__ void async_cp16(const unsigned short* g,
                                                  unsigned short* l) {
#if __has_builtin(__builtin_amdgcn_global_load_async_to_lds_b128)
  __builtin_amdgcn_global_load_async_to_lds_b128(
      (GAS v4i*)g, (LAS v4i*)l, 0, 0);
#else
  LAS void* lp = (LAS void*)l;
  unsigned int la; __builtin_memcpy(&la, &lp, 4);
  asm volatile("global_load_async_to_lds_b128 %0, %1, off"
               :: "v"(la), "v"(g) : "memory");
#endif
}

static __device__ __forceinline__ void wait_async0() {
#if __has_builtin(__builtin_amdgcn_s_wait_asynccnt)
  __builtin_amdgcn_s_wait_asynccnt(0);
#else
  asm volatile("s_wait_asynccnt 0x0" ::: "memory");
#endif
}

// ---------------------------------------------------------------------------
// Kernel 0: pack [x | h0] (fp32) into K-tiled bf16 A buffer: xt[kb][256][32]
// ---------------------------------------------------------------------------
__global__ void pack_xh(const float* __restrict__ x, const float* __restrict__ h0,
                        unsigned short* __restrict__ xt) {
  int idx = blockIdx.x * blockDim.x + threadIdx.x;   // over 256*5120
  int row = idx / KTOT;
  int col = idx - row * KTOT;
  float v = (col < IN_DIM) ? x[row * IN_DIM + col]
                           : h0[(size_t)row * HDIM + (col - IN_DIM)];
  int kb = col >> 5, kc = col & 31;
  xt[(size_t)kb * (BATCH * 32) + row * 32 + kc] = f2bf(v);
}

// ---------------------------------------------------------------------------
// GEMM: C[256 x Ntotal] = A(bf16,tiled) x W^T, W fp32 -> bf16 in-flight.
// 256 threads (8 waves), block tile M=256 x N=128, K step 32.
// Double-buffered padded LDS; A via async global->LDS; W prefetched in regs.
// ---------------------------------------------------------------------------
__global__ __launch_bounds__(256) void gemm_bf16(
    const unsigned short* __restrict__ At,
    const float* __restrict__ B0, int ld0,
    const float* __restrict__ B1, int ld1, int ksplit,
    float* __restrict__ C, int ldC, int kb_count)
{
  __shared__ __align__(16) unsigned short Al[2][BATCH * APAD];  // 2 x 20 KB
  __shared__ __align__(16) unsigned short Bl[2][128 * APAD];    // 2 x 10 KB

  const int tid    = threadIdx.x;
  const int lane   = tid & 31;
  const int w      = tid >> 5;
  const int wm     = w & 3;          // M quadrant (64 rows)
  const int wn     = w >> 2;         // N half (64 cols)
  const int lane16 = lane & 15;
  const int laneHi = lane >> 4;
  const int n0     = blockIdx.x * 128;

  // per-thread W-staging coordinates: row n0 + p*32 + r0, k chunk c4
  const int r0 = tid >> 3;           // 0..31
  const int c4 = tid & 7;            // float4 chunk within 32-wide K slice

  v8f acc[4][4];
#pragma unroll
  for (int i = 0; i < 4; ++i)
#pragma unroll
    for (int j = 0; j < 4; ++j)
      acc[i][j] = v8f{};

  float4 br[4];                      // W register prefetch

  auto stage_a = [&](unsigned short* dst, int kb) {
    const unsigned short* src = At + (size_t)kb * (BATCH * 32);
#pragma unroll
    for (int p = 0; p < 4; ++p) {
      int ci   = p * 256 + tid;      // 16B-chunk id: row = ci/4, quarter = ci&3
      int row  = ci >> 2, half = ci & 3;
      async_cp16(src + ci * 8, dst + row * APAD + half * 8);
    }
  };
  auto load_b = [&](int kb) {
    int k = kb * 32 + c4 * 4;
    const float* base; int ld;
    if (k < ksplit) { base = B0 + k;             ld = ld0; }
    else            { base = B1 + (k - ksplit);  ld = ld1; }
#pragma unroll
    for (int p = 0; p < 4; ++p) {
      int n = n0 + p * 32 + r0;
      br[p] = *(const float4*)(base + (size_t)n * ld);
    }
  };
  auto store_b = [&](unsigned short* dst) {
#pragma unroll
    for (int p = 0; p < 4; ++p) {
      v2bf ab = cvt2(br[p].x, br[p].y);
      v2bf cd = cvt2(br[p].z, br[p].w);
      v4bf v  = __builtin_shufflevector(ab, cd, 0, 1, 2, 3);
      *(v4bf*)(dst + (p * 32 + r0) * APAD + c4 * 4) = v;
    }
  };

  // ---- prologue: stage buffer 0 ----
  stage_a(Al[0], 0);
  load_b(0);
  store_b(Bl[0]);
  wait_async0();
  __syncthreads();

  // ---- main pipelined loop ----
  for (int kb = 0; kb < kb_count; ++kb) {
    const int cur = kb & 1, nxt = cur ^ 1;
    const bool have_next = (kb + 1 < kb_count);
    if (have_next) {
      stage_a(Al[nxt], kb + 1);    // async, overlaps compute below
      load_b(kb + 1);              // global->regs, overlaps compute below
    }

    // ---- compute from buffer `cur` ----
    const unsigned short* Ab = Al[cur];
    const unsigned short* Bb = Bl[cur];
    union F { v16bf v; uint4 q[2]; };
    F af, bw[4];
#pragma unroll
    for (int j = 0; j < 4; ++j) {
      int r = wn * 64 + j * 16 + lane16;
      const uint4* pB = (const uint4*)(Bb + r * APAD + laneHi * 8);
      bw[j].q[0] = pB[0];          // K laneHi*8 .. +7
      bw[j].q[1] = pB[2];          // K laneHi*8+16 .. +23
    }
#pragma unroll
    for (int i = 0; i < 4; ++i) {
      int r = wm * 64 + i * 16 + lane16;
      const uint4* pA = (const uint4*)(Ab + r * APAD + laneHi * 8);
      af.q[0] = pA[0];
      af.q[1] = pA[2];
#pragma unroll
      for (int j = 0; j < 4; ++j)
        acc[i][j] = __builtin_amdgcn_wmma_f32_16x16x32_bf16(
            false, af.v, false, bw[j].v, (short)0, acc[i][j], false, false);
    }

    if (have_next) {
      store_b(Bl[nxt]);            // waits W loads implicitly, cvt, LDS store
      wait_async0();               // our async A-chunks for `nxt` are in LDS
    }
    __syncthreads();               // buffer swap point
  }

  // ---- epilogue: C layout (VGPR r: M = laneHi*8 + r, N = lane16) ----
#pragma unroll
  for (int i = 0; i < 4; ++i) {
    int m0 = wm * 64 + i * 16 + laneHi * 8;
#pragma unroll
    for (int j = 0; j < 4; ++j) {
      int n = n0 + wn * 64 + j * 16 + lane16;
      float* dst = C + (size_t)m0 * ldC + n;
#pragma unroll
      for (int r = 0; r < 8; ++r) dst[(size_t)r * ldC] = acc[i][j][r];
    }
  }
}

// ---------------------------------------------------------------------------
// LSTM elementwise
// ---------------------------------------------------------------------------
__global__ void lstm_ew(const float* __restrict__ gates, const float* __restrict__ c0,
                        const float* __restrict__ b_ih, const float* __restrict__ b_hh,
                        float* __restrict__ h_out, float* __restrict__ c_out) {
  int idx = blockIdx.x * blockDim.x + threadIdx.x;     // over 256*4608
  int b = idx / HDIM, j = idx - b * HDIM;
  const float* g = gates + (size_t)b * G4;
  float ig = g[j]          + b_ih[j]          + b_hh[j];
  float fg = g[HDIM + j]   + b_ih[HDIM + j]   + b_hh[HDIM + j];
  float gg = g[2*HDIM + j] + b_ih[2*HDIM + j] + b_hh[2*HDIM + j];
  float og = g[3*HDIM + j] + b_ih[3*HDIM + j] + b_hh[3*HDIM + j];
  float si = 1.0f / (1.0f + __expf(-ig));
  float sf = 1.0f / (1.0f + __expf(-fg));
  float so = 1.0f / (1.0f + __expf(-og));
  float c  = sf * c0[idx] + si * tanhf(gg);
  float h  = so * tanhf(c);
  h_out[idx] = h;
  c_out[idx] = c;
}

// ---------------------------------------------------------------------------
// t[b][r] = sum_i dw2[b,r,i] * x[b,i],  dw2[b,r,i] = h_new[b, 2048 + r*512 + i]
// ---------------------------------------------------------------------------
__global__ void t_reduce(const float* __restrict__ x, const float* __restrict__ h_new,
                         float* __restrict__ t) {
  __shared__ float red[256];
  int b = blockIdx.x, tid = threadIdx.x;
  int r = tid >> 6, l = tid & 63;
  const float* xb = x + (size_t)b * IN_DIM;
  const float* d2 = h_new + (size_t)b * HDIM + OUT_DIM * RANK + r * IN_DIM;
  float s = 0.0f;
  for (int i = l; i < IN_DIM; i += 64) s += xb[i] * d2[i];
  red[tid] = s; __syncthreads();
  for (int off = 32; off >= 1; off >>= 1) {
    if (l < off) red[tid] += red[tid + off];
    __syncthreads();
  }
  if (l == 0) t[b * 4 + r] = red[tid];
}

// ---------------------------------------------------------------------------
// y[b,o] = (x@W0^T)[b,o] + sum_r h_new[b, o*4+r]*t[b,r] + h_new[b,4096+o] + bias0[o]
// ---------------------------------------------------------------------------
__global__ void y_final(const float* __restrict__ ygemm, const float* __restrict__ h_new,
                        const float* __restrict__ t, const float* __restrict__ bias0,
                        float* __restrict__ y) {
  int idx = blockIdx.x * blockDim.x + threadIdx.x;     // 256*512
  int b = idx >> 9, o = idx & 511;
  const float* hb = h_new + (size_t)b * HDIM;
  float acc = ygemm[idx] + hb[OUT_DIM * RANK + IN_DIM * RANK + o] + bias0[o];
  const float* tb = t + b * 4;
#pragma unroll
  for (int r = 0; r < 4; ++r) acc += hb[o * 4 + r] * tb[r];
  y[idx] = acc;
}

// ---------------------------------------------------------------------------
extern "C" void kernel_launch(void* const* d_in, const int* in_sizes, int n_in,
                              void* d_out, int out_size, void* d_ws, size_t ws_size,
                              hipStream_t stream) {
  const float* x     = (const float*)d_in[0];
  const float* h0    = (const float*)d_in[1];
  const float* c0    = (const float*)d_in[2];
  const float* W_ih  = (const float*)d_in[3];
  const float* W_hh  = (const float*)d_in[4];
  const float* b_ih  = (const float*)d_in[5];
  const float* b_hh  = (const float*)d_in[6];
  const float* W0    = (const float*)d_in[7];   // initial_weight (1,512,512)
  const float* bias0 = (const float*)d_in[8];   // initial_bias   (1,512)

  float* y     = (float*)d_out;
  float* h_out = y + BATCH * OUT_DIM;
  float* c_out = h_out + (size_t)BATCH * HDIM;

  char* ws = (char*)d_ws;
  unsigned short* xt = (unsigned short*)ws;                                  // 2.6 MB
  float* gates = (float*)(ws + (size_t)2 * BATCH * KTOT);                    // 18.9 MB
  float* ygemm = gates + (size_t)BATCH * G4;                                 // 0.5 MB
  float* tbuf  = ygemm + BATCH * OUT_DIM;                                    // 4 KB

  pack_xh<<<(BATCH * KTOT) / 256, 256, 0, stream>>>(x, h0, xt);

  // gates = [x|h0] @ [W_ih|W_hh]^T   (M=256, N=18432, K=5120)
  gemm_bf16<<<G4 / 128, 256, 0, stream>>>(xt, W_ih, IN_DIM, W_hh, HDIM, IN_DIM,
                                          gates, G4, KB_TOT);
  // ygemm = x @ W0^T                 (M=256, N=512, K=512)
  gemm_bf16<<<OUT_DIM / 128, 256, 0, stream>>>(xt, W0, IN_DIM, W0, IN_DIM, IN_DIM,
                                               ygemm, OUT_DIM, IN_DIM / 32);

  lstm_ew<<<(BATCH * HDIM) / 256, 256, 0, stream>>>(gates, c0, b_ih, b_hh, h_out, c_out);
  t_reduce<<<BATCH, 256, 0, stream>>>(x, h_out, tbuf);
  y_final<<<(BATCH * OUT_DIM) / 256, 256, 0, stream>>>(ygemm, h_out, tbuf, bias0, y);
}
